// HGTLayer_7842610283184
// MI455X (gfx1250) — compile-verified, matrix operand order checked by reading, not run on hardware
//
#include <hip/hip_runtime.h>

// ---------------------------------------------------------------------------
// HGT layer for MI455X (gfx1250, wave32).
// GEMMs run on v_wmma_f32_16x16x32_bf16 (fp32 -> bf16 truncation in LDS stage).
// Per-head 32x32 relation transforms are folded into the projection weights.
// Edge attention: one wave per (edge, head), DK=32 == wave32 lanes.
// ---------------------------------------------------------------------------

typedef __attribute__((ext_vector_type(16))) __bf16 v16bf;
typedef __attribute__((ext_vector_type(8)))  float  v8f;

#define DMODEL 256
#define NHEAD  8
#define DKDIM  32
#define BM     64
#define BN     128
#define KSLAB  64

__device__ __forceinline__ unsigned short f2bf(float f) {
    unsigned int u = __float_as_uint(f);
    u += 0x7fffu + ((u >> 16) & 1u);      // round-to-nearest-even truncation
    return (unsigned short)(u >> 16);
}

union FragBF { unsigned int u[8]; v16bf v; };

// ---------------------------------------------------------------------------
// C[M x 256] = A[M x 256] @ W[256 x 256]^T + bias      (EPI==0)
// out = alpha * (A@W^T + b) + (1-alpha) * Hin,  alpha = sigmoid(*skipp) (EPI==1)
// Block: 256 threads = 8 waves (4 M-waves x 2 N-waves). Wave tile 16x64.
// ---------------------------------------------------------------------------
template <int EPI>
__global__ __launch_bounds__(256)
void gemm_wmma(const float* __restrict__ A, const float* __restrict__ W,
               const float* __restrict__ bias, float* __restrict__ C, int M,
               const float* __restrict__ Hin, const float* __restrict__ skipp)
{
    __shared__ unsigned short sA[BM][KSLAB];   // 8 KB
    __shared__ unsigned short sB[BN][KSLAB];   // 16 KB

    const int tid   = threadIdx.x;
    const int lane  = tid & 31;
    const int wave  = tid >> 5;
    const int wm    = wave >> 1;     // 0..3 : M sub-tile
    const int wn    = wave & 1;      // 0..1 : N sub-tile
    const int m0    = blockIdx.x * BM;
    const int n0    = blockIdx.y * BN;
    const int khalf = lane >> 4;

    v8f acc[4] = {};

    for (int ks = 0; ks < DMODEL / KSLAB; ++ks) {
        const int k0 = ks * KSLAB;

        // Stage A tile (BM x KSLAB) as bf16; rows past M are zero.
        #pragma unroll
        for (int i = 0; i < (BM * KSLAB / 4) / 256; ++i) {
            int lin = i * 256 + tid;
            int row = lin >> 4;
            int c4  = lin & 15;
            float4 val = make_float4(0.f, 0.f, 0.f, 0.f);
            int gm = m0 + row;
            if (gm < M)
                val = *reinterpret_cast<const float4*>(A + (size_t)gm * DMODEL + k0 + c4 * 4);
            unsigned short* p = &sA[row][c4 * 4];
            p[0] = f2bf(val.x); p[1] = f2bf(val.y); p[2] = f2bf(val.z); p[3] = f2bf(val.w);
        }
        // Stage B tile (BN x KSLAB): row n of W is column n of B (W is out x in).
        #pragma unroll
        for (int i = 0; i < (BN * KSLAB / 4) / 256; ++i) {
            int lin = i * 256 + tid;
            int row = lin >> 4;
            int c4  = lin & 15;
            float4 val = *reinterpret_cast<const float4*>(W + (size_t)(n0 + row) * DMODEL + k0 + c4 * 4);
            unsigned short* p = &sB[row][c4 * 4];
            p[0] = f2bf(val.x); p[1] = f2bf(val.y); p[2] = f2bf(val.z); p[3] = f2bf(val.w);
        }
        __syncthreads();

        #pragma unroll
        for (int kk = 0; kk < KSLAB / 32; ++kk) {
            // A fragment: 16-bit A 16x32 layout (lanes 0-15 rows, lane-half = K group).
            FragBF a;
            const int am = wm * 16 + (lane & 15);
            #pragma unroll
            for (int j = 0; j < 8; ++j) {
                int pidx = (j < 4 ? j : j + 4) + (khalf << 2);
                a.u[j] = *reinterpret_cast<const unsigned int*>(&sA[am][kk * 32 + pidx * 2]);
            }
            #pragma unroll
            for (int t = 0; t < 4; ++t) {
                FragBF b;
                const int bn = wn * 64 + t * 16 + (lane & 15);
                #pragma unroll
                for (int j = 0; j < 8; ++j) {
                    int pidx = (j < 4 ? j : j + 4) + (khalf << 2);
                    b.u[j] = *reinterpret_cast<const unsigned int*>(&sB[bn][kk * 32 + pidx * 2]);
                }
                acc[t] = __builtin_amdgcn_wmma_f32_16x16x32_bf16(
                    false, a.v, false, b.v, (short)0, acc[t], false, false);
            }
        }
        __syncthreads();
    }

    float alpha = 0.f;
    if (EPI == 1) alpha = 1.f / (1.f + __expf(-skipp[0]));

    #pragma unroll
    for (int t = 0; t < 4; ++t) {
        int n_loc = wn * 64 + t * 16 + (lane & 15);
        int ng = n0 + n_loc;
        float bv = bias[ng];
        #pragma unroll
        for (int r = 0; r < 8; ++r) {
            int m_loc = wm * 16 + r + (khalf << 3);
            int mg = m0 + m_loc;
            if (mg < M) {
                float val = acc[t][r] + bv;
                if (EPI == 1) {
                    float hv = Hin[(size_t)mg * DMODEL + ng];
                    val = alpha * val + (1.f - alpha) * hv;
                }
                C[(size_t)mg * DMODEL + ng] = val;
            }
        }
    }
}

// ---------------------------------------------------------------------------
// Fold per-head relation transform into the projection weight:
//   W2[(h,j), d] = sum_i W[(h,i), d] * R[h,i,j];  b2 likewise.
// ---------------------------------------------------------------------------
__global__ void fuse_weight(const float* __restrict__ W, const float* __restrict__ R,
                            const float* __restrict__ b, float* __restrict__ W2,
                            float* __restrict__ b2)
{
    int t   = blockIdx.x * blockDim.x + threadIdx.x;   // 0..65535
    int row = t >> 8;                                   // h*32 + j
    int d   = t & 255;
    int h   = row >> 5;
    int j   = row & 31;
    const float* Rh = R + h * (DKDIM * DKDIM);
    float s = 0.f;
    #pragma unroll
    for (int i = 0; i < DKDIM; ++i)
        s += W[(size_t)(h * DKDIM + i) * DMODEL + d] * Rh[i * DKDIM + j];
    W2[(size_t)row * DMODEL + d] = s;
    if (d == 0) {
        float sb = 0.f;
        for (int i = 0; i < DKDIM; ++i) sb += b[h * DKDIM + i] * Rh[i * DKDIM + j];
        b2[row] = sb;
    }
}

__global__ void zero_u32(unsigned int* __restrict__ p, size_t n)
{
    size_t i = (size_t)blockIdx.x * blockDim.x + threadIdx.x;
    if (i < n) p[i] = 0u;
}

// One wave per (edge, head): dot(q[dst], k[src]) over 32 lanes, then
// segment-max via order-preserving uint atomicMax.
__global__ __launch_bounds__(256)
void edge_score(const float* __restrict__ q, const float* __restrict__ k,
                const int* __restrict__ src, const int* __restrict__ dst,
                const float* __restrict__ pri, int E_, float* __restrict__ score,
                unsigned int* __restrict__ mkey)
{
    int lane = threadIdx.x & 31;
    int gw   = (blockIdx.x * blockDim.x + threadIdx.x) >> 5;
    if (gw >= E_ * NHEAD) return;
    int e = gw >> 3, h = gw & 7;
    int s = src[e], d = dst[e];
    float p = q[(size_t)d * DMODEL + h * DKDIM + lane] *
              k[(size_t)s * DMODEL + h * DKDIM + lane];
    #pragma unroll
    for (int o = 16; o > 0; o >>= 1) p += __shfl_xor(p, o, 32);
    if (lane == 0) {
        float sc = p * pri[h] * 0.1767766952966369f;   // 1/sqrt(32)
        score[(size_t)e * NHEAD + h] = sc;
        unsigned int u   = __float_as_uint(sc);
        unsigned int key = (u & 0x80000000u) ? ~u : (u | 0x80000000u);
        atomicMax(&mkey[(size_t)d * NHEAD + h], key);
    }
}

__global__ void edge_expsum(const float* __restrict__ score, const int* __restrict__ dst,
                            const unsigned int* __restrict__ mkey, float* __restrict__ a,
                            float* __restrict__ ssum, int EH)
{
    int t = blockIdx.x * blockDim.x + threadIdx.x;
    if (t >= EH) return;
    int e = t >> 3, h = t & 7;
    int d = dst[e];
    unsigned int key = mkey[(size_t)d * NHEAD + h];
    unsigned int u   = (key & 0x80000000u) ? (key & 0x7fffffffu) : ~key;
    float m = __uint_as_float(u);
    if (((u >> 23) & 0xffu) == 0xffu) m = 0.f;   // non-finite -> 0 (reference)
    float av = __expf(score[t] - m);
    a[t] = av;
    atomicAdd(&ssum[(size_t)d * NHEAD + h], av);
}

__global__ __launch_bounds__(256)
void edge_scatter(const float* __restrict__ v, const float* __restrict__ a,
                  const float* __restrict__ ssum, const int* __restrict__ src,
                  const int* __restrict__ dst, int E_, float* __restrict__ agg)
{
    int lane = threadIdx.x & 31;
    int gw   = (blockIdx.x * blockDim.x + threadIdx.x) >> 5;
    if (gw >= E_ * NHEAD) return;
    int e = gw >> 3, h = gw & 7;
    int s = src[e], d = dst[e];
    float attn = a[(size_t)e * NHEAD + h] / ssum[(size_t)d * NHEAD + h];
    atomicAdd(&agg[(size_t)d * DMODEL + h * DKDIM + lane],
              v[(size_t)s * DMODEL + h * DKDIM + lane] * attn);
}

// ---------------------------------------------------------------------------

extern "C" void kernel_launch(void* const* d_in, const int* in_sizes, int n_in,
                              void* d_out, int out_size, void* d_ws, size_t ws_size,
                              hipStream_t stream)
{
    const float* h_author = (const float*)d_in[0];
    const float* h_paper  = (const float*)d_in[1];
    const float* k_w = (const float*)d_in[2];
    const float* k_b = (const float*)d_in[3];
    const float* q_w = (const float*)d_in[4];
    const float* q_b = (const float*)d_in[5];
    const float* v_w = (const float*)d_in[6];
    const float* v_b = (const float*)d_in[7];
    const float* a_w = (const float*)d_in[8];
    const float* a_b = (const float*)d_in[9];
    const float* rel_att = (const float*)d_in[10];
    const float* rel_msg = (const float*)d_in[11];
    const float* rel_pri = (const float*)d_in[12];
    const float* skip    = (const float*)d_in[13];
    const int* src_writes = (const int*)d_in[14];
    const int* dst_writes = (const int*)d_in[15];
    const int* src_cites  = (const int*)d_in[16];
    const int* dst_cites  = (const int*)d_in[17];
    const int* src_wb     = (const int*)d_in[18];
    const int* dst_wb     = (const int*)d_in[19];

    const int NA = in_sizes[0] / DMODEL;   // 20000
    const int NP = in_sizes[1] / DMODEL;   // 50000
    const int E  = in_sizes[14];           // 300000
    const size_t WSZ = (size_t)DMODEL * DMODEL;
    const size_t RSZ = (size_t)NHEAD * DKDIM * DKDIM;

    float* ws = (float*)d_ws;
    size_t off = 0;
    auto take = [&](size_t n) { float* p = ws + off; off += n; return p; };
    float* W2k  = take(WSZ);
    float* W2v  = take(WSZ);
    float* b2k  = take(DMODEL);
    float* b2v  = take(DMODEL);
    float* qbuf = take((size_t)NP * DMODEL);
    float* kbuf = take((size_t)NP * DMODEL);
    float* vbuf = take((size_t)NP * DMODEL);
    float* aggP = take((size_t)NP * DMODEL);
    float* aggA = take((size_t)NA * DMODEL);
    float* score = take((size_t)E * NHEAD);
    unsigned int* mkey = (unsigned int*)take((size_t)NP * NHEAD);
    float* ssum = take((size_t)NP * NHEAD);
    (void)ws_size; (void)n_in; (void)out_size;

    dim3 blk(256);
    auto zero = [&](void* p, size_t n) {
        zero_u32<<<dim3((unsigned)((n + 255) / 256)), blk, 0, stream>>>((unsigned int*)p, n);
    };
    auto gemm0 = [&](const float* A, const float* W, const float* b, float* C, int M) {
        dim3 g((M + BM - 1) / BM, DMODEL / BN);
        gemm_wmma<0><<<g, blk, 0, stream>>>(A, W, b, C, M, nullptr, nullptr);
    };
    auto gemm1 = [&](const float* A, const float* W, const float* b, float* C, int M,
                     const float* Hin, const float* sk) {
        dim3 g((M + BM - 1) / BM, DMODEL / BN);
        gemm_wmma<1><<<g, blk, 0, stream>>>(A, W, b, C, M, Hin, sk);
    };
    auto fuse = [&](const float* W, const float* R, const float* b, float* W2, float* b2) {
        fuse_weight<<<dim3(DMODEL * DMODEL / 256), blk, 0, stream>>>(W, R, b, W2, b2);
    };
    auto edges = [&](const float* q, const float* k, const float* v,
                     const int* src, const int* dst, int n_dst, const float* pri, float* agg) {
        zero(mkey, (size_t)n_dst * NHEAD);
        zero(ssum, (size_t)n_dst * NHEAD);
        edge_score<<<dim3(E), blk, 0, stream>>>(q, k, src, dst, pri, E, score, mkey);
        edge_expsum<<<dim3((E * NHEAD + 255) / 256), blk, 0, stream>>>(
            score, dst, mkey, score /*in-place*/, ssum, E * NHEAD);
        edge_scatter<<<dim3(E), blk, 0, stream>>>(v, score, ssum, src, dst, E, agg);
    };

    zero(aggP, (size_t)NP * DMODEL);
    zero(aggA, (size_t)NA * DMODEL);

    // q for papers (shared by etypes 0 and 1): q_w[1], q_b[1]
    gemm0(h_paper, q_w + WSZ, q_b + DMODEL, qbuf, NP);

    // etype 0: writes (author -> paper)
    fuse(k_w, rel_att, k_b, W2k, b2k);
    fuse(v_w, rel_msg, v_b, W2v, b2v);
    gemm0(h_author, W2k, b2k, kbuf, NA);
    gemm0(h_author, W2v, b2v, vbuf, NA);
    edges(qbuf, kbuf, vbuf, src_writes, dst_writes, NP, rel_pri, aggP);

    // etype 1: cites (paper -> paper)
    fuse(k_w + WSZ, rel_att + RSZ, k_b + DMODEL, W2k, b2k);
    fuse(v_w + WSZ, rel_msg + RSZ, v_b + DMODEL, W2v, b2v);
    gemm0(h_paper, W2k, b2k, kbuf, NP);
    gemm0(h_paper, W2v, b2v, vbuf, NP);
    edges(qbuf, kbuf, vbuf, src_cites, dst_cites, NP, rel_pri + NHEAD, aggP);

    // q for authors (etype 2): q_w[0], q_b[0]
    gemm0(h_author, q_w, q_b, qbuf, NA);

    // etype 2: written_by (paper -> author)
    fuse(k_w + WSZ, rel_att + 2 * RSZ, k_b + DMODEL, W2k, b2k);
    fuse(v_w + WSZ, rel_msg + 2 * RSZ, v_b + DMODEL, W2v, b2v);
    gemm0(h_paper, W2k, b2k, kbuf, NP);
    gemm0(h_paper, W2v, b2v, vbuf, NP);
    edges(qbuf, kbuf, vbuf, src_wb, dst_wb, NA, rel_pri + 2 * NHEAD, aggA);

    // outputs: (out_author, out_paper) concatenated
    float* out = (float*)d_out;
    gemm1(aggA, a_w, a_b, out, NA, h_author, skip);
    gemm1(aggP, a_w + WSZ, a_b + DMODEL, out + (size_t)NA * DMODEL, NP, h_paper, skip + 1);
}
